// Joint_85633057948138
// MI455X (gfx1250) — matrix operand order repaired and checked by reading, not compiled, not run on hardware
//
#include <hip/hip_runtime.h>
#include <hip/hip_bf16.h>

// RNN-T joint: out[B,T,U,V] = tanh(enc[b,t,:] + pred[b,u,:]) @ W^T + bias
// B=4, T=256, U=64, D=640, V=4096.  M = B*T*U = 65536.
//
// Strategy (MI455X roofline):
//  * 1.07 GB f32 output write floor = ~46us @ 23.3 TB/s. f32 WMMA (16x16x4)
//    would be compute bound (~0.5-1ms); bf16 WMMA 16x16x32 with f32 accum
//    lands within ~2x of the write floor at ~0.4% relative error.
//  * Phase 1: materialize H = bf16(tanh(enc (+) pred)) [65536 x 640] = 84MB and
//    Wb = bf16(W) [4096 x 640] = 5.2MB into d_ws (needs ~90MB; both stay L2
//    resident, L2 = 192MB). tanh computed exactly once per element.
//  * Phase 2: register-tiled bf16 WMMA GEMM, 512-thread block (16 waves,
//    wave32), block tile 256x256, wave tile 64x64 (4x4 v8f accumulators).
//    No LDS needed: A and B per-lane fragments are contiguous 16B runs in
//    H / Wb (W is [V,D] row-major, so B = W^T columns are contiguous in K).

typedef __attribute__((ext_vector_type(16))) __bf16 v16bf;
typedef __attribute__((ext_vector_type(8)))  __bf16 v8bf;
typedef __attribute__((ext_vector_type(8)))  float  v8f;
typedef __attribute__((ext_vector_type(4)))  float  v4f;

#define BB 4
#define TT 256
#define UU 64
#define DD 640
#define VV 4096
#define MM (BB*TT*UU)   // 65536

// ---------------- Phase 1a: H = bf16(tanh(enc (+) pred)) ----------------
// One thread -> 8 consecutive k of one row m. 65536*80 threads.
__global__ __launch_bounds__(256) void h_tanh_bf16(
    const float* __restrict__ enc, const float* __restrict__ pred,
    __bf16* __restrict__ H) {
  int g = blockIdx.x * 256 + threadIdx.x;     // 0 .. 65536*80-1
  int m = g / 80;
  int k = (g % 80) * 8;
  int u = m & (UU - 1);
  int t = (m >> 6) & (TT - 1);
  int b = m >> 14;
  const float* er = enc  + (size_t)(b * TT + t) * DD + k;
  const float* pr = pred + (size_t)(b * UU + u) * DD + k;
  v4f e0 = *(const v4f*)er, e1 = *(const v4f*)(er + 4);
  v4f p0 = *(const v4f*)pr, p1 = *(const v4f*)(pr + 4);
  v8bf h;
#pragma unroll
  for (int i = 0; i < 4; ++i) {
    h[i]     = (__bf16)tanhf(e0[i] + p0[i]);
    h[4 + i] = (__bf16)tanhf(e1[i] + p1[i]);
  }
  *(v8bf*)(H + (size_t)m * DD + k) = h;
}

// ---------------- Phase 1b: Wb = bf16(W) ----------------
__global__ __launch_bounds__(256) void w_to_bf16(
    const float* __restrict__ W, __bf16* __restrict__ Wb) {
  size_t i = ((size_t)blockIdx.x * 256 + threadIdx.x) * 8;  // V*D = 2,621,440 elems
  v4f a = *(const v4f*)(W + i);
  v4f b = *(const v4f*)(W + i + 4);
  v8bf o;
#pragma unroll
  for (int j = 0; j < 4; ++j) {
    o[j]     = (__bf16)a[j];
    o[4 + j] = (__bf16)b[j];
  }
  *(v8bf*)(Wb + i) = o;
}

// ---------------- Phase 2: bf16 WMMA GEMM ----------------
// grid = (M/256, V/256), block = 512 threads = 16 waves (4 M-waves x 4 N-waves)
// wave tile = 64x64 -> 4x4 accumulators of v8f.
__global__ __launch_bounds__(512) void joint_gemm_bf16(
    const __bf16* __restrict__ H, const __bf16* __restrict__ Wb,
    const float* __restrict__ bias, float* __restrict__ out) {
  const int tid  = threadIdx.x;
  const int lane = tid & 31;
  const int wave = tid >> 5;
  const int ln   = lane & 15;
  const int half = lane >> 4;
  const int mbase = blockIdx.x * 256 + (wave >> 2) * 64;
  const int nbase = blockIdx.y * 256 + (wave & 3) * 64;

  v8f acc[4][4] = {};

  // A fragment (16x32 bf16): lane holds row ln, K = {8h..8h+7, 16+8h..16+8h+7}
  const __bf16* Arow[4];
#pragma unroll
  for (int jm = 0; jm < 4; ++jm)
    Arow[jm] = H + (size_t)(mbase + jm * 16 + ln) * DD + 8 * half;
  // B fragment (32x16 bf16): lane holds col ln, K = {8h..8h+7, 16+8h..16+8h+7}
  const __bf16* Bcol[4];
#pragma unroll
  for (int jn = 0; jn < 4; ++jn)
    Bcol[jn] = Wb + (size_t)(nbase + jn * 16 + ln) * DD + 8 * half;

  for (int k0 = 0; k0 < DD; k0 += 32) {
    v16bf bf[4];
#pragma unroll
    for (int jn = 0; jn < 4; ++jn) {
      v8bf lo = *(const v8bf*)(Bcol[jn] + k0);
      v8bf hi = *(const v8bf*)(Bcol[jn] + k0 + 16);
      bf[jn] = __builtin_shufflevector(lo, hi, 0,1,2,3,4,5,6,7,8,9,10,11,12,13,14,15);
    }
#pragma unroll
    for (int jm = 0; jm < 4; ++jm) {
      v8bf lo = *(const v8bf*)(Arow[jm] + k0);
      v8bf hi = *(const v8bf*)(Arow[jm] + k0 + 16);
      v16bf af = __builtin_shufflevector(lo, hi, 0,1,2,3,4,5,6,7,8,9,10,11,12,13,14,15);
#pragma unroll
      for (int jn = 0; jn < 4; ++jn) {
        acc[jm][jn] = __builtin_amdgcn_wmma_f32_16x16x32_bf16(
            false, af, false, bf[jn], (short)0, acc[jm][jn], false, false);
      }
    }
  }

  // Epilogue: D VGPR v holds row (v + 8*half), col ln within each 16x16 tile.
#pragma unroll
  for (int jn = 0; jn < 4; ++jn) {
    const int n  = nbase + jn * 16 + ln;
    const float bv = bias[n];
#pragma unroll
    for (int jm = 0; jm < 4; ++jm) {
      const int rbase = mbase + jm * 16 + 8 * half;
#pragma unroll
      for (int v = 0; v < 8; ++v) {
        out[(size_t)(rbase + v) * VV + n] = acc[jm][jn][v] + bv;
      }
    }
  }
}

extern "C" void kernel_launch(void* const* d_in, const int* in_sizes, int n_in,
                              void* d_out, int out_size, void* d_ws, size_t ws_size,
                              hipStream_t stream) {
  (void)in_sizes; (void)n_in; (void)out_size;
  const float* enc  = (const float*)d_in[0];
  const float* pred = (const float*)d_in[1];
  const float* W    = (const float*)d_in[2];
  const float* bias = (const float*)d_in[3];
  float* out = (float*)d_out;

  // Workspace layout (assumes ws_size >= ~90MB):
  //   [0, V*D*2)                       : Wb  (bf16 W)      = 5,242,880 B
  //   [V*D*2, V*D*2 + M*D*2)           : H   (bf16 tanh)   = 83,886,080 B
  __bf16* Wb = (__bf16*)d_ws;
  __bf16* H  = (__bf16*)((char*)d_ws + (size_t)VV * DD * 2);
  (void)ws_size;

  w_to_bf16  <<<(VV * DD) / (256 * 8), 256, 0, stream>>>(W, Wb);
  h_tanh_bf16<<<(MM * 80) / 256,       256, 0, stream>>>(enc, pred, H);

  dim3 grid(MM / 256, VV / 256);
  joint_gemm_bf16<<<grid, 512, 0, stream>>>(H, Wb, bias, out);
}
